// CLFormer_62457414418995
// MI455X (gfx1250) — compile-verified, matrix operand order checked by reading, not run on hardware
//
#include <hip/hip_runtime.h>
#include <cmath>

#define BB   32
#define DD   128
#define SS   8192
#define HH   4
#define DH   32
#define NB   3
#define DOUTC 64

typedef __attribute__((ext_vector_type(16))) __bf16 v16bf;
typedef __attribute__((ext_vector_type(8)))  float  v8f;
typedef __attribute__((ext_vector_type(4)))  unsigned u32x4;
typedef __attribute__((ext_vector_type(8)))  int      i32x8;
typedef __attribute__((ext_vector_type(4)))  int      i32x4;

// K index inside a 16x32 bf16 WMMA fragment: half j of lane l covers
// K = (j&7) + (j>>3)*16 + (l>>4)*8   (ISA 7.12.2 16-bit A-matrix table)
__device__ __forceinline__ int kmap(int j, int laneHi) {
    return (j & 7) + ((j >> 3) << 4) + (laneHi << 3);
}

__device__ __forceinline__ float gelu_exact(float x) {
    return 0.5f * x * (1.0f + erff(x * 0.70710678118654752f));
}

// ---------------------------------------------------------------- TDM helper
// 2D tensor_load_to_lds: tile (tile_d0 x tile_d1) elements of (1<<ds_log) bytes,
// global row stride = stride0 elements. D# per ISA 8.3/8.4 (groups 2/3 NULL-ish,
// zero-filled; 2D tiles only need groups 0-1).
__device__ __forceinline__ void tdm_load_2d(unsigned lds_off, const void* gptr,
                                            unsigned ds_log,
                                            unsigned tile_d0, unsigned tile_d1,
                                            unsigned long stride0,
                                            unsigned tensor_d0, unsigned tensor_d1) {
    unsigned long ga = (unsigned long)gptr;
    u32x4 g0;
    g0[0] = 1u;                                              // count=1 (valid, user)
    g0[1] = lds_off;                                         // lds_addr
    g0[2] = (unsigned)(ga & 0xffffffffu);                    // global_addr[31:0]
    g0[3] = ((unsigned)(ga >> 32) & 0x01ffffffu) | (2u << 30); // ga[56:32] | type=2
    i32x8 g1;
    g1[0] = (int)(ds_log << 16);                             // wg_mask=0 | data_size
    g1[1] = (int)((tensor_d0 & 0xffffu) << 16);              // atomic_addr=0 | td0.lo
    g1[2] = (int)((tensor_d0 >> 16) | ((tensor_d1 & 0xffffu) << 16));
    g1[3] = (int)((tensor_d1 >> 16) | (tile_d0 << 16));      // td1.hi | tile_dim0
    g1[4] = (int)(tile_d1 & 0xffffu);                        // tile_dim1 | tile_dim2=0
    g1[5] = (int)(stride0 & 0xffffffffu);                    // dim0_stride[31:0]
    g1[6] = (int)((stride0 >> 32) & 0xffffu);                // dim0_stride[47:32]
    g1[7] = 0;
    i32x4 z4 = {0, 0, 0, 0};
#if __clang_major__ >= 23
    i32x8 z8 = {};
    __builtin_amdgcn_tensor_load_to_lds(g0, g1, z4, z4, z8, 0);
#else
    __builtin_amdgcn_tensor_load_to_lds(g0, g1, z4, z4, 0);
#endif
}

// ---------------------------------------------------------------- k_pe
// h[b][s][d] = x[b][d][s] + sinusoidal PE(s,d)
__global__ void k_pe(const float* __restrict__ x, float* __restrict__ h) {
    const float c = -logf(10000.0f) / (float)DD;
    const long total = (long)BB * SS * DD;
    for (long idx = (long)blockIdx.x * blockDim.x + threadIdx.x; idx < total;
         idx += (long)gridDim.x * blockDim.x) {
        int d = (int)(idx % DD);
        long t = idx / DD;
        int s = (int)(t % SS);
        int b = (int)(t / SS);
        int i2 = d & ~1;
        float ang = (float)s * __expf((float)i2 * c);
        float pe = (d & 1) ? __cosf(ang) : __sinf(ang);
        h[idx] = x[((long)b * DD + d) * SS + s] + pe;
    }
}

// ---------------------------------------------------------------- k_wconv
__global__ void k_wconv(const float* __restrict__ W1, const float* __restrict__ W2,
                        const float* __restrict__ Wf,
                        __bf16* __restrict__ W1b, __bf16* __restrict__ W2b,
                        __bf16* __restrict__ Wfb) {
    const int n1 = NB * DD * DD;
    for (int i = blockIdx.x * blockDim.x + threadIdx.x; i < n1;
         i += gridDim.x * blockDim.x) {
        W1b[i] = (__bf16)W1[i];
        W2b[i] = (__bf16)W2[i];
    }
    const int nf = DOUTC * DD;
    for (int i = blockIdx.x * blockDim.x + threadIdx.x; i < nf;
         i += gridDim.x * blockDim.x) {
        Wfb[i] = (__bf16)Wf[i];
    }
}

// ---------------------------------------------------------------- k_colstats
// Per column (b,d): max over s and sum_s exp(h - max)   (softmax over seq axis)
__global__ void k_colstats(const float* __restrict__ h,
                           float* __restrict__ cmax, float* __restrict__ csum) {
    __shared__ float red[256];
    const int col = blockIdx.x;            // b*DD + d
    const int b = col / DD, d = col % DD;
    const float* base = h + (long)b * SS * DD + d;
    const int t = threadIdx.x;

    float m = -1e30f;
    for (int s = t; s < SS; s += 256) m = fmaxf(m, base[(long)s * DD]);
    red[t] = m; __syncthreads();
    for (int o = 128; o > 0; o >>= 1) {
        if (t < o) red[t] = fmaxf(red[t], red[t + o]);
        __syncthreads();
    }
    m = red[0]; __syncthreads();

    float sm = 0.f;
    for (int s = t; s < SS; s += 256) sm += __expf(base[(long)s * DD] - m);
    red[t] = sm; __syncthreads();
    for (int o = 128; o > 0; o >>= 1) {
        if (t < o) red[t] += red[t + o];
        __syncthreads();
    }
    if (t == 0) { cmax[col] = m; csum[col] = red[0]; }
}

// ---------------------------------------------------------------- k_ctx
// ctx[b][hh][d][e] = sum_s k_sm[s][d] * qh[s][e]   (32x32 per (b,hh))
// One workgroup (8 waves) per (b,hh); waves split the S axis in 32-row chunks.
// The 256x32 strided h tile is staged into LDS by the Tensor Data Mover.
__global__ void k_ctx(const float* __restrict__ h, const float* __restrict__ cmax,
                      const float* __restrict__ csum, float* __restrict__ ctx) {
    extern __shared__ char sm_dyn[];
    float* tile = (float*)sm_dyn;            // 256 x 32 fp32  (32 KB) @ LDS 0
    float* part = tile + 256 * DH;           // 8 x 32 x 32    (32 KB) @ LDS 32768

    const int bh = blockIdx.x;
    const int b = bh / HH, hh = bh % HH;
    const int t = threadIdx.x;
    const int wave = t >> 5, lane = t & 31;
    const int laneHi = lane >> 4, ln = lane & 15;

    // per-lane seq-softmax constants for the two A row-tiles (d = row of A)
    const float mx0 = cmax[b * DD + hh * DH + ln];
    const float is0 = 1.0f / csum[b * DD + hh * DH + ln];
    const float mx1 = cmax[b * DD + hh * DH + 16 + ln];
    const float is1 = 1.0f / csum[b * DD + hh * DH + 16 + ln];

    v8f acc[2][2] = {};
    const float* hb = h + (long)b * SS * DD + hh * DH;

    for (int s0 = 0; s0 < SS; s0 += 256) {
        __syncthreads();                       // previous tile fully consumed
        if (t < 32) {                          // wave 0 drives the TDM
            // 2D tile: 32 floats x 256 rows, row stride = DD floats
            tdm_load_2d(0u, hb + (long)s0 * DD, /*ds_log=*/2u,
                        /*tile_d0=*/DH, /*tile_d1=*/256u,
                        /*stride0=*/(unsigned long)DD,
                        /*tensor_d0=*/DH, /*tensor_d1=*/256u);
            __builtin_amdgcn_s_wait_tensorcnt(0);
        }
        __syncthreads();
        const float* tw = tile + wave * 32 * DH;   // this wave's 32 s-rows
        #pragma unroll
        for (int tm = 0; tm < 2; ++tm) {
            const float mx = tm ? mx1 : mx0;
            const float is = tm ? is1 : is0;
            const int m = tm * 16 + ln;
            v16bf a;
            #pragma unroll
            for (int j = 0; j < 16; ++j) {
                int k = kmap(j, laneHi);
                a[j] = (__bf16)(__expf(tw[k * DH + m] - mx) * is);
            }
            #pragma unroll
            for (int tn = 0; tn < 2; ++tn) {
                const int n = tn * 16 + ln;
                v16bf bv;
                #pragma unroll
                for (int j = 0; j < 16; ++j) {
                    int k = kmap(j, laneHi);
                    bv[j] = (__bf16)tw[k * DH + n];
                }
                acc[tm][tn] = __builtin_amdgcn_wmma_f32_16x16x32_bf16(
                    false, a, false, bv, (short)0, acc[tm][tn], false, false);
            }
        }
    }

    __syncthreads();
    #pragma unroll
    for (int tm = 0; tm < 2; ++tm)
        #pragma unroll
        for (int tn = 0; tn < 2; ++tn)
            #pragma unroll
            for (int v = 0; v < 8; ++v) {
                int m = tm * 16 + v + 8 * laneHi;
                int n = tn * 16 + ln;
                part[(wave * 32 + m) * 32 + n] = acc[tm][tn][v];
            }
    __syncthreads();
    for (int e = t; e < 32 * 32; e += 256) {
        float s = 0.f;
        #pragma unroll
        for (int w = 0; w < 8; ++w) s += part[(w * 32 + (e >> 5)) * 32 + (e & 31)];
        ctx[(long)bh * 32 * 32 + e] = s;
    }
}

// 128x128x128 bf16 WMMA GEMM helper: wave computes 16 rows x 128 cols.
// A, W in LDS; W stored row-major as (out=n, in=k): B[k][n] = W[n*DD + k].
__device__ __forceinline__ void gemm128(const __bf16* A, const __bf16* W,
                                        int wave, int lane, v8f* acc) {
    const int laneHi = lane >> 4, ln = lane & 15;
    const v8f vz = {};
    #pragma unroll
    for (int nt = 0; nt < 8; ++nt) acc[nt] = vz;
    #pragma unroll
    for (int kt = 0; kt < 4; ++kt) {
        const int m = wave * 16 + ln;
        v16bf a;
        #pragma unroll
        for (int j = 0; j < 16; ++j)
            a[j] = A[m * DD + kt * 32 + kmap(j, laneHi)];
        #pragma unroll
        for (int nt = 0; nt < 8; ++nt) {
            const int n = nt * 16 + ln;
            v16bf bv;
            #pragma unroll
            for (int j = 0; j < 16; ++j)
                bv[j] = W[n * DD + kt * 32 + kmap(j, laneHi)];
            acc[nt] = __builtin_amdgcn_wmma_f32_16x16x32_bf16(
                false, a, false, bv, (short)0, acc[nt], false, false);
        }
    }
}

// ---------------------------------------------------------------- k_attffn
// For a 128-row tile of (b): att = q_sm @ ctx (fp32 VALU), then
// h_new = att + gelu(gelu(att@W1^T+b1)@W2^T+b2) via two WMMA GEMMs.
// W1/W2 (32 KB bf16 each) staged into LDS by the Tensor Data Mover.
__global__ void k_attffn(float* __restrict__ h, const float* __restrict__ ctxg,
                         const __bf16* __restrict__ W1b, const float* __restrict__ b1,
                         const __bf16* __restrict__ W2b, const float* __restrict__ b2) {
    extern __shared__ char sm_dyn[];
    float*  attf = (float*)sm_dyn;                 // 128x128 fp32  (64 KB) @ 0
    float*  ctxs = attf + DD * DD;                 // 4x32x32 fp32  (16 KB) @ 65536
    __bf16* abf  = (__bf16*)(ctxs + HH * DH * DH); // 128x128 bf16  (32 KB) @ 81920
    __bf16* wl   = abf + DD * DD;                  // 128x128 bf16  (32 KB) @ 114688
    const unsigned WL_OFF = (unsigned)(DD * DD * 4 + HH * DH * DH * 4 + DD * DD * 2);

    const int s0 = blockIdx.x * DD;
    const int b  = blockIdx.y;
    const int t  = threadIdx.x;
    const int wave = t >> 5, lane = t & 31;
    const int laneHi = lane >> 4, ln = lane & 15;

    // stage ctx (all heads); TDM pulls W1; prefetch W2 block toward L2
    for (int i = t; i < HH * DH * DH; i += 256) ctxs[i] = ctxg[(long)b * HH * DH * DH + i];
    if (t < 32) {
        tdm_load_2d(WL_OFF, W1b, /*ds_log=*/1u,
                    /*tile_d0=*/(unsigned)(DD * DD), /*tile_d1=*/1u,
                    /*stride0=*/(unsigned long)(DD * DD),
                    /*tensor_d0=*/(unsigned)(DD * DD), /*tensor_d1=*/1u);
        __builtin_amdgcn_s_wait_tensorcnt(0);
    }
    if (t < 128) __builtin_prefetch(W2b + t * 128, 0, 1);  // global_prefetch_b8
    __syncthreads();

    // Phase 1: att rows (head-dim softmax + 32x32 matvec against ctx)
    for (int p = t; p < DD * HH; p += 256) {       // 512 (row,head) pairs
        const int r = p >> 2, hh = p & 3;
        const float* row = h + ((long)b * SS + s0 + r) * DD + hh * DH;
        float v[32];
        float mx = -1e30f;
        #pragma unroll
        for (int c = 0; c < 32; ++c) { v[c] = row[c]; mx = fmaxf(mx, v[c]); }
        float sm = 0.f;
        #pragma unroll
        for (int c = 0; c < 32; ++c) { v[c] = __expf(v[c] - mx); sm += v[c]; }
        const float inv = 1.0f / sm;
        const float* cx = ctxs + hh * DH * DH;
        for (int e = 0; e < 32; ++e) {
            float a = 0.f;
            #pragma unroll
            for (int c = 0; c < 32; ++c) a += v[c] * cx[c * DH + e];
            a *= inv;
            attf[r * DD + hh * DH + e] = a;
            abf [r * DD + hh * DH + e] = (__bf16)a;
        }
    }
    __syncthreads();

    v8f acc[8];

    // Phase 2: f1 = gelu(att @ W1^T + b1), bf16 result back into abf (own rows)
    gemm128(abf, wl, wave, lane, acc);
    #pragma unroll
    for (int nt = 0; nt < 8; ++nt) {
        const int n = nt * 16 + ln;
        const float bias = b1[n];
        #pragma unroll
        for (int v = 0; v < 8; ++v) {
            const int m = wave * 16 + v + 8 * laneHi;
            abf[m * DD + n] = (__bf16)gelu_exact(acc[nt][v] + bias);
        }
    }
    __syncthreads();                 // all waves done reading W1 from wl
    if (t < 32) {
        tdm_load_2d(WL_OFF, W2b, 1u, (unsigned)(DD * DD), 1u,
                    (unsigned long)(DD * DD), (unsigned)(DD * DD), 1u);
        __builtin_amdgcn_s_wait_tensorcnt(0);
    }
    __syncthreads();

    // Phase 3: h_new = att + gelu(f1 @ W2^T + b2)
    gemm128(abf, wl, wave, lane, acc);
    #pragma unroll
    for (int nt = 0; nt < 8; ++nt) {
        const int n = nt * 16 + ln;
        const float bias = b2[n];
        #pragma unroll
        for (int v = 0; v < 8; ++v) {
            const int m = wave * 16 + v + 8 * laneHi;
            const float g = gelu_exact(acc[nt][v] + bias);
            h[((long)b * SS + s0 + m) * DD + n] = attf[m * DD + n] + g;
        }
    }
}

// ---------------------------------------------------------------- k_final
// out[b][o][s] = sum_d h[b][s][d] * Wf[o][d] + bf[o]
// h tile (fp32) and Wf (bf16) staged into LDS by the Tensor Data Mover.
__global__ void k_final(const float* __restrict__ h, const __bf16* __restrict__ Wfb,
                        const float* __restrict__ bfv, float* __restrict__ out) {
    extern __shared__ char sm_dyn[];
    float*  hf  = (float*)sm_dyn;                  // 128x128 fp32 (64 KB) @ 0
    __bf16* wfl = (__bf16*)(hf + DD * DD);         // 64x128 bf16  (16 KB) @ 65536
    const unsigned WF_OFF = (unsigned)(DD * DD * 4);

    const int s0 = blockIdx.x * DD;
    const int b  = blockIdx.y;
    const int t  = threadIdx.x;
    const int wave = t >> 5, lane = t & 31;
    const int laneHi = lane >> 4, ln = lane & 15;

    if (t < 32) {
        // contiguous 128x128 fp32 tile of h (rows s0..s0+127)
        tdm_load_2d(0u, h + ((long)b * SS + s0) * DD, /*ds_log=*/2u,
                    /*tile_d0=*/DD, /*tile_d1=*/DD,
                    /*stride0=*/(unsigned long)DD,
                    /*tensor_d0=*/DD, /*tensor_d1=*/DD);
        // Wf block, contiguous bf16
        tdm_load_2d(WF_OFF, Wfb, 1u, (unsigned)(DOUTC * DD), 1u,
                    (unsigned long)(DOUTC * DD), (unsigned)(DOUTC * DD), 1u);
        __builtin_amdgcn_s_wait_tensorcnt(0);
    }
    __syncthreads();

    const v8f vz = {};
    v8f acc[4];
    #pragma unroll
    for (int nt = 0; nt < 4; ++nt) acc[nt] = vz;
    #pragma unroll
    for (int kt = 0; kt < 4; ++kt) {
        const int m = wave * 16 + ln;
        v16bf a;
        #pragma unroll
        for (int j = 0; j < 16; ++j)
            a[j] = (__bf16)hf[m * DD + kt * 32 + kmap(j, laneHi)];
        #pragma unroll
        for (int nt = 0; nt < 4; ++nt) {
            const int n = nt * 16 + ln;
            v16bf bv;
            #pragma unroll
            for (int j = 0; j < 16; ++j)
                bv[j] = wfl[n * DD + kt * 32 + kmap(j, laneHi)];
            acc[nt] = __builtin_amdgcn_wmma_f32_16x16x32_bf16(
                false, a, false, bv, (short)0, acc[nt], false, false);
        }
    }
    #pragma unroll
    for (int nt = 0; nt < 4; ++nt) {
        const int n = nt * 16 + ln;
        const float bias = bfv[n];
        #pragma unroll
        for (int v = 0; v < 8; ++v) {
            const int m = wave * 16 + v + 8 * laneHi;
            out[((long)b * DOUTC + n) * SS + s0 + m] = acc[nt][v] + bias;
        }
    }
}

extern "C" void kernel_launch(void* const* d_in, const int* in_sizes, int n_in,
                              void* d_out, int out_size, void* d_ws, size_t ws_size,
                              hipStream_t stream) {
    const float* x   = (const float*)d_in[0];
    const float* W1  = (const float*)d_in[1];
    const float* b1  = (const float*)d_in[2];
    const float* W2  = (const float*)d_in[3];
    const float* b2  = (const float*)d_in[4];
    const float* Wf  = (const float*)d_in[5];
    const float* bfv = (const float*)d_in[6];
    float* out = (float*)d_out;

    char* ws = (char*)d_ws;
    size_t off = 0;
    float* h    = (float*)(ws + off); off += (size_t)BB * SS * DD * 4;   // 134 MB
    float* cmax = (float*)(ws + off); off += (size_t)BB * DD * 4;
    float* csum = (float*)(ws + off); off += (size_t)BB * DD * 4;
    float* ctx  = (float*)(ws + off); off += (size_t)BB * HH * DH * DH * 4;
    __bf16* W1b = (__bf16*)(ws + off); off += (size_t)NB * DD * DD * 2;
    __bf16* W2b = (__bf16*)(ws + off); off += (size_t)NB * DD * DD * 2;
    __bf16* Wfb = (__bf16*)(ws + off); off += (size_t)DOUTC * DD * 2;
    (void)off; (void)ws_size; (void)in_sizes; (void)n_in; (void)out_size;

    k_pe<<<16384, 256, 0, stream>>>(x, h);
    k_wconv<<<256, 256, 0, stream>>>(W1, W2, Wf, W1b, W2b, Wfb);

    const size_t lds_ctx    = (size_t)(256 * DH + 8 * DH * DH) * 4;              // 64 KB
    const size_t lds_attffn = (size_t)DD * DD * 4 + (size_t)HH * DH * DH * 4
                            + (size_t)DD * DD * 2 * 2;                           // 144 KB
    const size_t lds_final  = (size_t)DD * DD * 4 + (size_t)DOUTC * DD * 2;      // 80 KB

    for (int i = 0; i < NB; ++i) {
        k_colstats<<<BB * DD, 256, 0, stream>>>(h, cmax, csum);
        k_ctx<<<BB * HH, 256, lds_ctx, stream>>>(h, cmax, csum, ctx);
        k_attffn<<<dim3(SS / DD, BB), 256, lds_attffn, stream>>>(
            h, ctx, W1b + (size_t)i * DD * DD, b1 + i * DD,
            W2b + (size_t)i * DD * DD, b2 + i * DD);
    }
    k_final<<<dim3(SS / DD, BB), 256, lds_final, stream>>>(h, Wfb, bfv, out);
}
